// MultiheadAttention_83116207112193
// MI455X (gfx1250) — matrix operand order repaired and checked by reading, not compiled, not run on hardware
//
#include <hip/hip_runtime.h>

// ---------------------------------------------------------------------------
// MI455X / gfx1250 multi-head attention forward.
// bf16 WMMA (V_WMMA_F32_16X16X32_BF16), f32 accumulate + f32 softmax.
// All matrix fragments are assembled with 16-byte vector loads so the inner
// loops are wmma + b128 loads, not per-element packing.
// ---------------------------------------------------------------------------

typedef __bf16 bf16_t;
typedef __attribute__((ext_vector_type(16))) __bf16 v16bf;
typedef __attribute__((ext_vector_type(8)))  float  v8f;

#define EMBED 512
#define SEQ   2048
#define BATCH 4
#define HEADS 8
#define HDIM  64

static __device__ inline v8f wmma_bf16(v16bf a, v16bf b, v8f c) {
  // 8 args: (neg_a, A, neg_b, B, c_mod, C, reuse_a, reuse_b)
  return __builtin_amdgcn_wmma_f32_16x16x32_bf16(false, a, false, b, (short)0, c,
                                                 false, false);
}

// Build a v16bf from two 16-byte chunks (ISA layouts are always 8- or
// 16-element contiguous runs per lane).  Works for global and LDS pointers.
static __device__ inline v16bf ld16(const bf16_t* p0, const bf16_t* p1) {
  union { v16bf v; uint4 q[2]; } u;
  u.q[0] = *(const uint4*)p0;
  u.q[1] = *(const uint4*)p1;
  return u.v;
}

// A fragment (16x32 bf16): lane l: row = l&15, e[0..15] = K kb..kb+15,
// kb = (l>>4)*16.  p points at (row, kb).
static __device__ inline v16bf ld_afrag(const bf16_t* p) { return ld16(p, p + 8); }
// B fragment (32x16 bf16), column-per-lane with K contiguous: 32 bytes.
static __device__ inline v16bf ld_bfrag(const bf16_t* p) { return ld16(p, p + 8); }

// ---------------------------------------------------------------------------
// Pre-pass 0a: x (f32) -> xb (bf16), coalesced, 4 elems/thread.
// ---------------------------------------------------------------------------
__global__ __launch_bounds__(256)
void cvt_x_kernel(const float* __restrict__ x, bf16_t* __restrict__ xb) {
  const size_t i = ((size_t)blockIdx.x * 256 + threadIdx.x) * 4;
  const float4 t = *(const float4*)(x + i);
  union { uint2 u; __bf16 e[4]; } pk;
  pk.e[0] = (__bf16)t.x; pk.e[1] = (__bf16)t.y;
  pk.e[2] = (__bf16)t.z; pk.e[3] = (__bf16)t.w;
  *(uint2*)(xb + i) = pk.u;
}

// ---------------------------------------------------------------------------
// Pre-pass 0b: w [K][N] f32 -> wt [N][K] bf16 (so GEMM B-fragments are
// contiguous 32B loads).  grid = (N/256, K).
// ---------------------------------------------------------------------------
__global__ __launch_bounds__(256)
void transpose_w_kernel(const float* __restrict__ w, bf16_t* __restrict__ wt,
                        int K, int N) {
  const int n = blockIdx.x * 256 + threadIdx.x;
  const int k = blockIdx.y;
  wt[(size_t)n * K + k] = (__bf16)w[(size_t)k * N + n];
}

// ---------------------------------------------------------------------------
// Kernel 1: qkv = xb @ w_in^T(+bias); q pre-scaled.  M=8192, N=1536, K=512.
// Block 128 thr (4 waves), block tile 128(M) x 64(N); wave tile 32 x 64.
// No LDS, no barriers: A and B fragments are direct global b128 loads
// (weights come from the bf16-transposed copy, L2-resident).
// q,k -> [B,H,S,D] bf16;  v -> [B,H,D,S] bf16 (transposed, packed stores).
// ---------------------------------------------------------------------------
__global__ __launch_bounds__(128)
void qkv_proj_kernel(const bf16_t* __restrict__ xb, const bf16_t* __restrict__ wtin,
                     const float* __restrict__ b_in,
                     bf16_t* __restrict__ qbuf, bf16_t* __restrict__ kbuf,
                     bf16_t* __restrict__ vT) {
  const int tid = threadIdx.x, wave = tid >> 5, lane = tid & 31;
  const int l15 = lane & 15, hi = lane >> 4;
  const int M0 = blockIdx.y * 128 + wave * 32;
  const int N0 = blockIdx.x * 64;

  v8f acc[2][4] = {};

  for (int kk = 0; kk < EMBED; kk += 32) {
    v16bf a[2];
#pragma unroll
    for (int m = 0; m < 2; ++m)
      a[m] = ld_afrag(xb + (size_t)(M0 + m * 16 + l15) * EMBED + kk + hi * 16);
#pragma unroll
    for (int t = 0; t < 4; ++t) {
      const v16bf b =
          ld_bfrag(wtin + (size_t)(N0 + t * 16 + l15) * EMBED + kk + hi * 16);
      acc[0][t] = wmma_bf16(a[0], b, acc[0][t]);
      acc[1][t] = wmma_bf16(a[1], b, acc[1][t]);
    }
  }

#pragma unroll
  for (int m = 0; m < 2; ++m) {
    const int row0 = M0 + m * 16 + hi * 8;          // 8 consecutive rows
    const int bb = row0 >> 11, s0 = row0 & 2047;    // never crosses batch
#pragma unroll
    for (int t = 0; t < 4; ++t) {
      const int gcol = N0 + t * 16 + l15;
      const int sel = gcol >> 9;                    // 0=q, 1=k, 2=v
      const int e512 = gcol & 511;
      const int h = e512 >> 6, d = e512 & 63;
      const float bias = b_in[gcol];
      if (sel == 2) {
        // v transposed: d-major, 8 consecutive s -> one 16B store
        union { uint4 q; __bf16 e[8]; } pk;
#pragma unroll
        for (int r = 0; r < 8; ++r) pk.e[r] = (__bf16)(acc[m][t][r] + bias);
        *(uint4*)&vT[(((size_t)bb * HEADS + h) * HDIM + d) * SEQ + s0] = pk.q;
      } else {
        bf16_t* dst = sel ? kbuf : qbuf;
        const float scale = sel ? 1.0f : 0.125f;    // HEAD_DIM^-0.5 on q
#pragma unroll
        for (int r = 0; r < 8; ++r) {
          const float val = (acc[m][t][r] + bias) * scale;
          dst[(((size_t)bb * HEADS + h) * SEQ + (s0 + r)) * HDIM + d] = (__bf16)val;
        }
      }
    }
  }
}

// ---------------------------------------------------------------------------
// Kernel 2: fused attention for one (b,h) x 16-query tile.  Block 256 (8 waves).
//  P1: scores(16x2048) = q @ k^T -> LDS f32           (2 WMMA / key tile)
//  P2: f32 row softmax (x4-vectorized); writes f32 weights to global attn
//      AND a bf16 copy into LDS for phase 3
//  P3: o(16x64) = attn @ v, K split over wave halves, LDS reduce
// Dynamic LDS: 128K f32 scores + 64K bf16 weights + 8K reduce = 200 KB.
// ---------------------------------------------------------------------------
__global__ __launch_bounds__(256)
void attn_kernel(const bf16_t* __restrict__ qbuf, const bf16_t* __restrict__ kbuf,
                 const bf16_t* __restrict__ vT, float* __restrict__ attn_out,
                 bf16_t* __restrict__ obuf) {
  extern __shared__ char smem[];
  float*  sc  = (float*)smem;                         // [16][2048] f32 scores
  bf16_t* scb = (bf16_t*)(smem + 16 * SEQ * 4);       // [16][2048] bf16 weights
  float*  red = (float*)(smem + 16 * SEQ * 6);        // [2][16][64] partials

  const int tid = threadIdx.x, wave = tid >> 5, lane = tid & 31;
  const int l15 = lane & 15, hi = lane >> 4;
  const int bh = blockIdx.y;                          // b*H + h
  const int q0 = blockIdx.x * 16;

  const bf16_t* qp = qbuf + (size_t)bh * SEQ * HDIM;
  const bf16_t* kp = kbuf + (size_t)bh * SEQ * HDIM;
  const bf16_t* vp = vT + (size_t)bh * HDIM * SEQ;

  // q A-fragments, two K=32 chunks over D=64 (q pre-scaled in kernel 1)
  v16bf aq[2];
#pragma unroll
  for (int c = 0; c < 2; ++c)
    aq[c] = ld_afrag(qp + (size_t)(q0 + l15) * HDIM + c * 32 + hi * 16);

  // ---- Phase 1: scores, 16 key tiles per wave ----
  for (int it = 0; it < 16; ++it) {
    const int kt = wave + 8 * it;
    if (it + 1 < 16)
      __builtin_prefetch(kp + (size_t)(kt + 8) * 16 * HDIM + lane * 32, 0, 0);
    v8f acc = {};
#pragma unroll
    for (int c = 0; c < 2; ++c) {
      const v16bf b =
          ld_bfrag(kp + (size_t)(kt * 16 + l15) * HDIM + c * 32 + hi * 16);
      acc = wmma_bf16(aq[c], b, acc);
    }
#pragma unroll
    for (int r = 0; r < 8; ++r)
      sc[(hi * 8 + r) * SEQ + kt * 16 + l15] = acc[r];
  }
  __syncthreads();

  // ---- Phase 2: softmax, 2 rows per wave, x4 vectorized ----
  for (int rr = 0; rr < 2; ++rr) {
    const int row = wave * 2 + rr;
    float* sr = sc + (size_t)row * SEQ;
    bf16_t* sb = scb + (size_t)row * SEQ;
    float m = -3.4e38f;
    for (int j = lane * 4; j < SEQ; j += 128) {
      const float4 t = *(const float4*)(sr + j);
      m = fmaxf(m, fmaxf(fmaxf(t.x, t.y), fmaxf(t.z, t.w)));
    }
#pragma unroll
    for (int o = 16; o > 0; o >>= 1) m = fmaxf(m, __shfl_xor(m, o, 32));
    float sum = 0.f;
    for (int j = lane * 4; j < SEQ; j += 128) {
      float4 t = *(const float4*)(sr + j);
      t.x = __expf(t.x - m); t.y = __expf(t.y - m);
      t.z = __expf(t.z - m); t.w = __expf(t.w - m);
      sum += (t.x + t.y) + (t.z + t.w);
      *(float4*)(sr + j) = t;
    }
#pragma unroll
    for (int o = 16; o > 0; o >>= 1) sum += __shfl_xor(sum, o, 32);
    const float inv = 1.0f / sum;
    float* ar = attn_out + ((size_t)bh * SEQ + (q0 + row)) * SEQ;
    for (int j = lane * 4; j < SEQ; j += 128) {
      float4 t = *(const float4*)(sr + j);
      t.x *= inv; t.y *= inv; t.z *= inv; t.w *= inv;
      *(float4*)(ar + j) = t;                  // mandatory attn output (512 MB)
      union { uint2 u; __bf16 e[4]; } pk;      // bf16 copy for phase 3
      pk.e[0] = (__bf16)t.x; pk.e[1] = (__bf16)t.y;
      pk.e[2] = (__bf16)t.z; pk.e[3] = (__bf16)t.w;
      *(uint2*)(sb + j) = pk.u;
    }
  }
  __syncthreads();

  // ---- Phase 3: o = attn @ v.  wave -> (d-tile = wave&3, K-half = wave>>2) ----
  const int dt = wave & 3, half = wave >> 2;
  v8f oacc = {};
  for (int ks = 0; ks < 64; ++ks) {
    const int kk = (half * 64 + ks) * 32;      // 32-key K chunk
    const v16bf a = ld_afrag(scb + (size_t)l15 * SEQ + kk + hi * 16);
    const v16bf b =
        ld_bfrag(vp + (size_t)(dt * 16 + l15) * SEQ + kk + hi * 16);
    oacc = wmma_bf16(a, b, oacc);
  }
#pragma unroll
  for (int r = 0; r < 8; ++r)
    red[half * 1024 + (hi * 8 + r) * 64 + dt * 16 + l15] = oacc[r];
  __syncthreads();

  // Reduce K-halves; packed bf16 store of merged-heads row [B*S, E].
  {
    const int b_ = bh >> 3, h_ = bh & 7;
    const int i = tid * 4;                     // 256 thr x 4 = 1024 elems
    const int row = i >> 6, d = i & 63;
    union { uint2 u; __bf16 e[4]; } pk;
#pragma unroll
    for (int j = 0; j < 4; ++j) pk.e[j] = (__bf16)(red[i + j] + red[1024 + i + j]);
    *(uint2*)&obuf[((size_t)b_ * SEQ + q0 + row) * EMBED + h_ * HDIM + d] = pk.u;
  }
}

// ---------------------------------------------------------------------------
// Kernel 3: out = o_heads @ w_out^T(+bias).  M=8192, N=512, K=512.
// Same barrier-free structure as kernel 1; f32 epilogue into d_out.
// ---------------------------------------------------------------------------
__global__ __launch_bounds__(128)
void out_proj_kernel(const bf16_t* __restrict__ obuf, const bf16_t* __restrict__ wtout,
                     const float* __restrict__ b_out, float* __restrict__ out) {
  const int tid = threadIdx.x, wave = tid >> 5, lane = tid & 31;
  const int l15 = lane & 15, hi = lane >> 4;
  const int M0 = blockIdx.y * 128 + wave * 32;
  const int N0 = blockIdx.x * 64;

  v8f acc[2][4] = {};

  for (int kk = 0; kk < EMBED; kk += 32) {
    v16bf a[2];
#pragma unroll
    for (int m = 0; m < 2; ++m)
      a[m] = ld_afrag(obuf + (size_t)(M0 + m * 16 + l15) * EMBED + kk + hi * 16);
#pragma unroll
    for (int t = 0; t < 4; ++t) {
      const v16bf b =
          ld_bfrag(wtout + (size_t)(N0 + t * 16 + l15) * EMBED + kk + hi * 16);
      acc[0][t] = wmma_bf16(a[0], b, acc[0][t]);
      acc[1][t] = wmma_bf16(a[1], b, acc[1][t]);
    }
  }

#pragma unroll
  for (int m = 0; m < 2; ++m) {
    const int row0 = M0 + m * 16 + hi * 8;
#pragma unroll
    for (int t = 0; t < 4; ++t) {
      const int gcol = N0 + t * 16 + l15;
      const float bias = b_out[gcol];
#pragma unroll
      for (int r = 0; r < 8; ++r)
        out[(size_t)(row0 + r) * EMBED + gcol] = acc[m][t][r] + bias;
    }
  }
}

// ---------------------------------------------------------------------------
// Launcher
// ---------------------------------------------------------------------------
extern "C" void kernel_launch(void* const* d_in, const int* in_sizes, int n_in,
                              void* d_out, int out_size, void* d_ws, size_t ws_size,
                              hipStream_t stream) {
  (void)in_sizes; (void)n_in; (void)out_size; (void)ws_size;

  const float* x     = (const float*)d_in[0];   // [4, 2048, 512]
  const float* w_in  = (const float*)d_in[1];   // [512, 1536]
  const float* b_in  = (const float*)d_in[2];   // [1536]
  const float* w_out = (const float*)d_in[3];   // [512, 512]
  const float* b_out = (const float*)d_in[4];   // [512]

  float* out  = (float*)d_out;                                 // o: 4*2048*512
  float* attn = out + (size_t)BATCH * SEQ * EMBED;             // [32,2048,2048]

  const size_t elems = (size_t)BATCH * SEQ * EMBED;            // 4M
  bf16_t* xb    = (bf16_t*)d_ws;                               // 8 MB
  bf16_t* wtin  = xb + elems;                                  // 1.5 MB [1536][512]
  bf16_t* wtout = wtin + (size_t)3 * EMBED * EMBED;            // 0.5 MB [512][512]
  bf16_t* qbuf  = wtout + (size_t)EMBED * EMBED;               // 8 MB  [B,H,S,D]
  bf16_t* kbuf  = qbuf + elems;                                // 8 MB  [B,H,S,D]
  bf16_t* vT    = kbuf + elems;                                // 8 MB  [B,H,D,S]
  bf16_t* obuf  = vT + elems;                                  // 8 MB  [B*S, E]

  // P0: dtype conversion / weight transposes (once, ~30 MB traffic)
  cvt_x_kernel<<<(elems / 4) / 256, 256, 0, stream>>>(x, xb);
  transpose_w_kernel<<<dim3((3 * EMBED) / 256, EMBED), 256, 0, stream>>>(
      w_in, wtin, EMBED, 3 * EMBED);
  transpose_w_kernel<<<dim3(EMBED / 256, EMBED), 256, 0, stream>>>(
      w_out, wtout, EMBED, EMBED);

  // K1: QKV projection  (grid: N/64 x M/128)
  qkv_proj_kernel<<<dim3((3 * EMBED) / 64, (BATCH * SEQ) / 128), 128, 0, stream>>>(
      xb, wtin, b_in, qbuf, kbuf, vT);

  // K2: fused attention (grid: S/16 query tiles x B*H), 200 KB dynamic LDS
  const size_t smem_bytes = (size_t)16 * SEQ * 6 + 2 * 16 * 64 * sizeof(float);
  attn_kernel<<<dim3(SEQ / 16, BATCH * HEADS), 256, smem_bytes, stream>>>(
      qbuf, kbuf, vT, attn, obuf);

  // K3: output projection (grid: N/64 x M/128)
  out_proj_kernel<<<dim3(EMBED / 64, (BATCH * SEQ) / 128), 128, 0, stream>>>(
      obuf, wtout, b_out, out);
}
